// Attention_45681272160463
// MI455X (gfx1250) — matrix operand order repaired and checked by reading, not compiled, not run on hardware
//
#include <hip/hip_runtime.h>

typedef __attribute__((ext_vector_type(16))) _Float16 v16h;
typedef __attribute__((ext_vector_type(8)))  _Float16 v8h;
typedef __attribute__((ext_vector_type(4)))  _Float16 v4h;
typedef __attribute__((ext_vector_type(8)))  float    v8f;

#define WMMA_F32_F16(a, b, c) \
  __builtin_amdgcn_wmma_f32_16x16x32_f16(false, (a), false, (b), (short)0, (c), false, false)

union AFrag { v16h v; v8h h[2]; };

static __device__ __forceinline__ v16h load16(const _Float16* p) {
  // 16 contiguous halfs, 16B-aligned: two ds_load_b128
  AFrag f;
  f.h[0] = *(const v8h*)p;
  f.h[1] = *(const v8h*)(p + 8);
  return f.v;
}

static __device__ __forceinline__ v8h cvt8(const float* p, float s) {
  v8h r;
#pragma unroll
  for (int j = 0; j < 8; ++j) r[j] = (_Float16)(p[j] * s);
  return r;
}

// DPP16 reductions across each 16-lane half (pure VALU, no LDS traffic).
static __device__ __forceinline__ float row16_max(float x) {
  x = fmaxf(x, __int_as_float(__builtin_amdgcn_update_dpp(0, __float_as_int(x), 0x0B1, 0xF, 0xF, true)));
  x = fmaxf(x, __int_as_float(__builtin_amdgcn_update_dpp(0, __float_as_int(x), 0x04E, 0xF, 0xF, true)));
  x = fmaxf(x, __int_as_float(__builtin_amdgcn_update_dpp(0, __float_as_int(x), 0x141, 0xF, 0xF, true)));
  x = fmaxf(x, __int_as_float(__builtin_amdgcn_update_dpp(0, __float_as_int(x), 0x140, 0xF, 0xF, true)));
  return x;
}
static __device__ __forceinline__ float row16_sum(float x) {
  x += __int_as_float(__builtin_amdgcn_update_dpp(0, __float_as_int(x), 0x0B1, 0xF, 0xF, true));
  x += __int_as_float(__builtin_amdgcn_update_dpp(0, __float_as_int(x), 0x04E, 0xF, 0xF, true));
  x += __int_as_float(__builtin_amdgcn_update_dpp(0, __float_as_int(x), 0x141, 0xF, 0xF, true));
  x += __int_as_float(__builtin_amdgcn_update_dpp(0, __float_as_int(x), 0x140, 0xF, 0xF, true));
  return x;
}

// B=4, H=16, S=2048, D=64. Flash attention: 64-query tile per block (4 wave32 x 16 rows),
// 128-key inner blocks, f16 WMMA / f32 accum, online softmax in exp2 domain.
#define ATT_S 2048
#define ATT_D 64
#define KB    128   // keys per inner block
#define NSUB  (KB / 16)
#define KSTR  72    // kS row stride (halfs): 144B, 16B-aligned, spreads banks
#define VSTR  136   // vS/pS row stride (halfs): 272B, 16B-aligned

// (128, 4): min 4 waves/EU -> cap 256 VGPRs/wave; 2 blocks co-reside per CU.
__global__ __launch_bounds__(128, 4) void Attention_45681272160463_kernel(
    const float* __restrict__ Q, const float* __restrict__ K,
    const float* __restrict__ V, float* __restrict__ O) {
  __shared__ __align__(16) _Float16 kS[KB * KSTR];       // K block, row-major [key][d]
  __shared__ __align__(16) _Float16 vS[ATT_D * VSTR];    // V block, transposed [d][key]
  __shared__ __align__(16) _Float16 pS[4 * 16 * VSTR];   // per-wave P scratch [m][key]

  const int tid  = threadIdx.x;
  const int wave = tid >> 5;
  const int lane = tid & 31;
  const int half = lane >> 4;
  const int l16  = lane & 15;

  const int nqb = ATT_S / 64;
  const int bh  = blockIdx.x / nqb;
  const int qb  = blockIdx.x % nqb;

  const size_t base  = (size_t)bh * ATT_S * ATT_D;
  const int    qrow0 = qb * 64 + wave * 16;

  // ---- Persistent Q A-fragments (two 16x32 halves of 16x64), pre-scaled ----
  // A-layout: lane m=l16; elem e -> k=half*16+e
  const float qscale = 0.125f * 1.44269504088896340736f;  // 1/sqrt(64) * log2(e)
  const float* qrow = Q + base + (size_t)(qrow0 + l16) * ATT_D;
  AFrag aq0, aq1;
  aq0.h[0] = cvt8(qrow +      half * 16, qscale);
  aq0.h[1] = cvt8(qrow +  8 + half * 16, qscale);
  aq1.h[0] = cvt8(qrow + 32 + half * 16, qscale);
  aq1.h[1] = cvt8(qrow + 40 + half * 16, qscale);

  // ---- Online-softmax state (C/D layout: row m = r + 8*half, col = l16) ----
  float mrun[8], lrun[8];   // lrun = per-LANE partial sums; reduced in epilogue
  v8f o[4];
#pragma unroll
  for (int r = 0; r < 8; ++r) { mrun[r] = -1e30f; lrun[r] = 0.0f; }
#pragma unroll
  for (int c = 0; c < 4; ++c) o[c] = (v8f){};

  _Float16* pw = pS + wave * 16 * VSTR;

  for (int kb = 0; kb < ATT_S / KB; ++kb) {
    // ---- Cooperative fill: K row-major f16 (packed b64 stores), V transposed f16 ----
    const float4* kg = (const float4*)(K + base + (size_t)kb * KB * ATT_D);
    const float4* vg = (const float4*)(V + base + (size_t)kb * KB * ATT_D);
#pragma unroll
    for (int i = 0; i < 16; ++i) {
      int idx4 = i * 128 + tid;           // 2048 float4 = 128x64 elements
      int key  = idx4 >> 4;
      int d0   = (idx4 & 15) << 2;
      float4 kk = kg[idx4];
      float4 vv = vg[idx4];
      v4h kp;
      kp[0] = (_Float16)kk.x; kp[1] = (_Float16)kk.y;
      kp[2] = (_Float16)kk.z; kp[3] = (_Float16)kk.w;
      *(v4h*)&kS[key * KSTR + d0] = kp;   // one ds_store_b64
      vS[(d0 + 0) * VSTR + key] = (_Float16)vv.x;
      vS[(d0 + 1) * VSTR + key] = (_Float16)vv.y;
      vS[(d0 + 2) * VSTR + key] = (_Float16)vv.z;
      vS[(d0 + 3) * VSTR + key] = (_Float16)vv.w;
    }
    __syncthreads();

    // Prefetch next K/V block into cache while computing (global_prefetch_b8)
    if (kb + 1 < ATT_S / KB) {
      const char* nk = (const char*)(K + base + (size_t)(kb + 1) * KB * ATT_D);
      const char* nv = (const char*)(V + base + (size_t)(kb + 1) * KB * ATT_D);
#pragma unroll
      for (int j = 0; j < 2; ++j) {
        __builtin_prefetch(nk + tid * 256 + j * 128, 0, 1);
        __builtin_prefetch(nv + tid * 256 + j * 128, 0, 1);
      }
    }

    // ---- S = Q * K^T : 8 key-subtiles x (2 WMMAs over D=64) ----
    // B-layout (32x16): lane n=l16, elem e -> kcontr = half*16+e  => contiguous d run
    v8f s[NSUB];
#pragma unroll
    for (int sub = 0; sub < NSUB; ++sub) {
      const _Float16* kp = kS + (sub * 16 + l16) * KSTR + half * 16;
      v16h b0 = load16(kp);        // d = 0..31 across halves
      v16h b1 = load16(kp + 32);   // d = 32..63
      v8f acc = (v8f){};
      acc = WMMA_F32_F16(aq0.v, b0, acc);
      acc = WMMA_F32_F16(aq1.v, b1, acc);
      s[sub] = acc;
    }

    // ---- Fused per-row online softmax: max -> alpha -> rescale -> exp/P-store ----
#pragma unroll
    for (int r = 0; r < 8; ++r) {
      float mx = s[0][r];
#pragma unroll
      for (int sub = 1; sub < NSUB; ++sub) mx = fmaxf(mx, s[sub][r]);
      mx = row16_max(mx);
      const float mnew  = fmaxf(mrun[r], mx);
      const float alpha = exp2f(mrun[r] - mnew);
      mrun[r] = mnew;
#pragma unroll
      for (int c = 0; c < 4; ++c) o[c][r] *= alpha;
      const int m = r + half * 8;
      float rs = 0.0f;
#pragma unroll
      for (int sub = 0; sub < NSUB; ++sub) {
        float p = exp2f(s[sub][r] - mnew);
        rs += p;
        pw[m * VSTR + sub * 16 + l16] = (_Float16)p;
      }
      lrun[r] = lrun[r] * alpha + rs;   // lane-partial; reduced at end
    }

    // ---- O += P * V : j (32-key group) outer so only one P frag is live ----
    const _Float16* pr = pw + l16 * VSTR;
#pragma unroll
    for (int j = 0; j < 4; ++j) {
      AFrag pf;
      pf.h[0] = *(const v8h*)(pr + j * 32 +     half * 16);
      pf.h[1] = *(const v8h*)(pr + j * 32 + 8 + half * 16);
#pragma unroll
      for (int c = 0; c < 4; ++c) {
        const _Float16* vp = vS + (c * 16 + l16) * VSTR + half * 16;
        v16h bv = load16(vp + j * 32);   // keys j*32 .. j*32+31
        o[c] = WMMA_F32_F16(pf.v, bv, o[c]);
      }
    }
    __syncthreads();   // protect kS/vS before next iteration's fill
  }

  // ---- Final row-sum reduction, normalize, write out ----
  float inv[8];
#pragma unroll
  for (int r = 0; r < 8; ++r) inv[r] = 1.0f / row16_sum(lrun[r]);
  float* op = O + base + (size_t)qrow0 * ATT_D;
#pragma unroll
  for (int c = 0; c < 4; ++c)
#pragma unroll
    for (int r = 0; r < 8; ++r)
      op[(size_t)(r + half * 8) * ATT_D + c * 16 + l16] = o[c][r] * inv[r];
}

extern "C" void kernel_launch(void* const* d_in, const int* in_sizes, int n_in,
                              void* d_out, int out_size, void* d_ws, size_t ws_size,
                              hipStream_t stream) {
  (void)in_sizes; (void)n_in; (void)out_size; (void)d_ws; (void)ws_size;
  const float* Q = (const float*)d_in[0];
  const float* K = (const float*)d_in[1];
  const float* V = (const float*)d_in[2];
  float* O = (float*)d_out;
  // grid: (B*H) * (S/64) = 64 * 32 = 2048 blocks of 128 threads (4 wave32)
  dim3 grid(4 * 16 * (ATT_S / 64));
  dim3 block(128);
  Attention_45681272160463_kernel<<<grid, block, 0, stream>>>(Q, K, V, O);
}
